// RNNPool_50672024158484
// MI455X (gfx1250) — matrix operand order repaired and checked by reading, not compiled, not run on hardware
//
#include <hip/hip_runtime.h>

// ---------------------------------------------------------------------------
// RNNPool (FastGRNN row/col + shared bidirectional) fused kernel for gfx1250.
// Single kernel, input read once from HBM (256 MB -> ~11 us floor @23.3 TB/s),
// all matmuls via v_wmma_f32_16x16x32_bf16 (fp32 accumulate).
// Elementwise gates use hardware v_tanh_f32 (branch-free).
// ---------------------------------------------------------------------------

typedef __attribute__((ext_vector_type(16))) __bf16 v16bf;
typedef __attribute__((ext_vector_type(8)))  __bf16 v8bf;
typedef __attribute__((ext_vector_type(8)))  float  v8f;

#define H_DIM 64
#define LDSX_RCSTRIDE 72     // bf16 elems per (r,c) row of one batch (64 + pad)
#define LDSX_BSTRIDE  4616   // bf16 elems per batch: 64*72 + 8 pad (bank step 4)
#define HST_MSTRIDE   72     // bf16 elems per hidden row (64 + pad)
#define HST_SLOT      (16*HST_MSTRIDE)
#define NUM_HST_SLOTS 20     // 16 phase-1 waves + 4 phase-2 waves
#define SMEM_ELEMS    (16*LDSX_BSTRIDE + NUM_HST_SLOTS*HST_SLOT)
#define SMEM_BYTES    (SMEM_ELEMS*2)

#if defined(__has_builtin)
#if __has_builtin(__builtin_amdgcn_tanhf)
#define HAVE_HW_TANH 1
#endif
#endif

static __device__ __forceinline__ float tanh_fast(float x) {
#ifdef HAVE_HW_TANH
    return __builtin_amdgcn_tanhf(x);          // v_tanh_f32
#else
    const float e = __expf(-2.0f * x);         // branch-free fallback
    return (1.0f - e) * __builtin_amdgcn_rcpf(1.0f + e);
#endif
}

static __device__ __forceinline__ float sigm_fast(float x) {
    return 0.5f + 0.5f * tanh_fast(0.5f * x);
}

// B-fragment (32x16, K x N) of a row-major [64 x 64] f32 weight matrix,
// converted to bf16.  ISA layout: lane n = l&15, half = l>>4;
// VGPR v (g=v>>2, j=v&3) holds K = g*16 + half*8 + 2j, +1.
static __device__ __forceinline__ v16bf load_b_frag(const float* __restrict__ Wg,
                                                    int kbase, int nt, int lane) {
    const int nl = lane & 15, half = lane >> 4;
    v16bf r;
#pragma unroll
    for (int v = 0; v < 8; ++v) {
        const int k = kbase + (v >> 2) * 16 + half * 8 + 2 * (v & 3);
        r[2 * v]     = (__bf16)Wg[(k)     * 64 + nt * 16 + nl];
        r[2 * v + 1] = (__bf16)Wg[(k + 1) * 64 + nt * 16 + nl];
    }
    return r;
}

// A-fragment (16x32, M x K) from bf16 LDS where K is contiguous per M-row.
// ISA 16-bit A layout: lane m=l&15, half=l>>4; VGPR v (g=v>>2, j=v&3) holds
// K = g*16 + half*8 + 2j, +1  ==> fragment = concat of two aligned 16B chunks
// at k0 = kbase + half*8 and k0 + 16.  Two ds_load_b128 per fragment.
static __device__ __forceinline__ v16bf load_a_frag(const __bf16* rowbase,
                                                    int row_stride, int kbase, int lane) {
    const int m = lane & 15, half = lane >> 4;
    const __bf16* p = rowbase + m * row_stride + kbase + half * 8;
    const v8bf c0 = *(const v8bf*)(p);
    const v8bf c1 = *(const v8bf*)(p + 16);
    return __builtin_shufflevector(c0, c1, 0, 1, 2, 3, 4, 5, 6, 7,
                                           8, 9, 10, 11, 12, 13, 14, 15);
}

// One FastGRNN step:
//   pre = x_t @ W + h @ U   (4 k-slices x 4 N-tiles, runs of 4 indep. WMMAs)
//   z = sigmoid(pre+bg); c = tanh(pre+bu)
//   h = z*h + (szt*(1-z) + snu)*c  (fp32, C-layout), bf16 mirror -> hslot
static __device__ __forceinline__ void fastgrnn_step(
    const v16bf& xa0, const v16bf& xa1,
    __bf16* hslot,
    const v16bf (*Wf)[4], const v16bf (*Uf)[4],
    v8f* h, const float* bgl, const float* bul,
    float szt, float snu, int lane)
{
    const v16bf ha0 = load_a_frag(hslot, HST_MSTRIDE, 0,  lane);
    const v16bf ha1 = load_a_frag(hslot, HST_MSTRIDE, 32, lane);
    const int nl = lane & 15, half = lane >> 4;

    v8f acc[4];
#pragma unroll
    for (int nt = 0; nt < 4; ++nt)
        acc[nt] = (v8f){0.f, 0.f, 0.f, 0.f, 0.f, 0.f, 0.f, 0.f};
#pragma unroll
    for (int nt = 0; nt < 4; ++nt)
        acc[nt] = __builtin_amdgcn_wmma_f32_16x16x32_bf16(false, xa0, false, Wf[0][nt],
                                                          (short)0, acc[nt], false, false);
#pragma unroll
    for (int nt = 0; nt < 4; ++nt)
        acc[nt] = __builtin_amdgcn_wmma_f32_16x16x32_bf16(false, xa1, false, Wf[1][nt],
                                                          (short)0, acc[nt], false, false);
#pragma unroll
    for (int nt = 0; nt < 4; ++nt)
        acc[nt] = __builtin_amdgcn_wmma_f32_16x16x32_bf16(false, ha0, false, Uf[0][nt],
                                                          (short)0, acc[nt], false, false);
#pragma unroll
    for (int nt = 0; nt < 4; ++nt)
        acc[nt] = __builtin_amdgcn_wmma_f32_16x16x32_bf16(false, ha1, false, Uf[1][nt],
                                                          (short)0, acc[nt], false, false);

#pragma unroll
    for (int nt = 0; nt < 4; ++nt) {
#pragma unroll
        for (int e = 0; e < 8; ++e) {
            const float pre = acc[nt][e];
            const float z  = sigm_fast(pre + bgl[nt]);
            const float cc = tanh_fast(pre + bul[nt]);
            const float hn = z * h[nt][e] + (szt * (1.0f - z) + snu) * cc;
            h[nt][e] = hn;
            hslot[(half * 8 + e) * HST_MSTRIDE + nt * 16 + nl] = (__bf16)hn;
        }
    }
}

// Load h0 (f32, C-layout registers) and mirror bf16 into the staging slot.
static __device__ __forceinline__ void init_h(const float* __restrict__ h0p,
                                              v8f* h, __bf16* hslot, int lane)
{
    const int nl = lane & 15, half = lane >> 4;
#pragma unroll
    for (int nt = 0; nt < 4; ++nt) {
#pragma unroll
        for (int v = 0; v < 8; ++v) {
            const int m = half * 8 + v;
            const float hv = h0p[(size_t)m * H_DIM + nt * 16 + nl];
            h[nt][v] = hv;
            hslot[m * HST_MSTRIDE + nt * 16 + nl] = (__bf16)hv;
        }
    }
}

extern __shared__ __align__(16) char smem_raw[];

__global__ __launch_bounds__(512, 1)
void rnnpool_fused(const float* __restrict__ inp,
                   const float* __restrict__ W1, const float* __restrict__ U1,
                   const float* __restrict__ bg1, const float* __restrict__ bu1,
                   const float* __restrict__ zeta1, const float* __restrict__ nu1,
                   const float* __restrict__ W2, const float* __restrict__ U2,
                   const float* __restrict__ bg2, const float* __restrict__ bu2,
                   const float* __restrict__ zeta2, const float* __restrict__ nu2,
                   const float* __restrict__ h0_row, const float* __restrict__ h0_col,
                   const float* __restrict__ h0_bd_row, const float* __restrict__ h0_bd_col,
                   float* __restrict__ out, int B)
{
    __bf16* lds_x = (__bf16*)smem_raw;                 // [16 batches][64 rc][64 d] bf16
    __bf16* hst   = lds_x + 16 * LDSX_BSTRIDE;         // 20 slots of [16][72] bf16

    const int tid  = threadIdx.x;
    const int wv   = tid >> 5;
    const int lane = tid & 31;
    const int b0   = blockIdx.x * 16;

    // ---- stage input tile transposed: (bi, d, rc) -> lds_x[bi][rc][d] bf16 ----
    {
        const float4* gin = (const float4*)(inp + (size_t)b0 * 4096);
#pragma unroll 4
        for (int i = tid * 4; i < 16 * 4096; i += 512 * 4) {
            const float4 v4 = gin[i >> 2];
            const int bi = i >> 12;
            const int d  = (i >> 6) & 63;
            const int rc = i & 63;                     // rc .. rc+3 (c innermost)
            __bf16* p = lds_x + bi * LDSX_BSTRIDE + rc * LDSX_RCSTRIDE + d;
            p[0 * LDSX_RCSTRIDE] = (__bf16)v4.x;
            p[1 * LDSX_RCSTRIDE] = (__bf16)v4.y;
            p[2 * LDSX_RCSTRIDE] = (__bf16)v4.z;
            p[3 * LDSX_RCSTRIDE] = (__bf16)v4.w;
        }
    }
    __syncthreads();

    v16bf Wf[2][4], Uf[2][4];
    v8f   h[4];
    float bgl[4], bul[4];
    const int nl = lane & 15;

    // =========================== Phase 1: row & col passes ==================
    {
#pragma unroll
        for (int kt = 0; kt < 2; ++kt)
#pragma unroll
            for (int nt = 0; nt < 4; ++nt) {
                Wf[kt][nt] = load_b_frag(W1, kt * 32, nt, lane);
                Uf[kt][nt] = load_b_frag(U1, kt * 32, nt, lane);
            }
#pragma unroll
        for (int nt = 0; nt < 4; ++nt) {
            bgl[nt] = bg1[nt * 16 + nl];
            bul[nt] = bu1[nt * 16 + nl];
        }
        const float szt = sigm_fast(zeta1[0]);
        const float snu = sigm_fast(nu1[0]);

        const bool isRow = (wv < 8);
        const int  fixed = isRow ? wv : (wv - 8);      // c for row pass, r for col pass
        const float* h0p = (isRow ? h0_row : h0_col)
                         + ((size_t)fixed * B + b0) * H_DIM;
        __bf16* hslot = hst + wv * HST_SLOT;

        init_h(h0p, h, hslot, lane);

#pragma unroll
        for (int t = 0; t < 8; ++t) {
            const int rc = isRow ? (t * 8 + fixed) : (fixed * 8 + t);
            const __bf16* xrow = lds_x + rc * LDSX_RCSTRIDE;   // + m*LDSX_BSTRIDE
            const v16bf xa0 = load_a_frag(xrow, LDSX_BSTRIDE, 0,  lane);
            const v16bf xa1 = load_a_frag(xrow, LDSX_BSTRIDE, 32, lane);
            fastgrnn_step(xa0, xa1, hslot, Wf, Uf, h, bgl, bul, szt, snu, lane);
        }
        // final h of wave w now lives in hst slot w == states_{r|c} for this tile
    }
    __syncthreads();

    // ====================== Phase 2: shared bidirectional ===================
    if (wv < 4) {
#pragma unroll
        for (int kt = 0; kt < 2; ++kt)
#pragma unroll
            for (int nt = 0; nt < 4; ++nt) {
                Wf[kt][nt] = load_b_frag(W2, kt * 32, nt, lane);
                Uf[kt][nt] = load_b_frag(U2, kt * 32, nt, lane);
            }
#pragma unroll
        for (int nt = 0; nt < 4; ++nt) {
            bgl[nt] = bg2[nt * 16 + nl];
            bul[nt] = bu2[nt * 16 + nl];
        }
        const float szt = sigm_fast(zeta2[0]);
        const float snu = sigm_fast(nu2[0]);

        const bool rowp = (wv < 2);
        const int  dir  = wv & 1;                      // 0 = forward, 1 = backward
        const float* h0p = (rowp ? h0_bd_row : h0_bd_col)
                         + ((size_t)dir * B + b0) * H_DIM;
        __bf16* hslot = hst + (16 + wv) * HST_SLOT;

        init_h(h0p, h, hslot, lane);

#pragma unroll
        for (int t = 0; t < 8; ++t) {
            const int step = dir ? (7 - t) : t;
            const int slot = (rowp ? 0 : 8) + step;
            const __bf16* xs = hst + slot * HST_SLOT;
            const v16bf xa0 = load_a_frag(xs, HST_MSTRIDE, 0,  lane);
            const v16bf xa1 = load_a_frag(xs, HST_MSTRIDE, 32, lane);
            fastgrnn_step(xa0, xa1, hslot, Wf, Uf, h, bgl, bul, szt, snu, lane);
        }

        // out[b, cbase + n]: [rows_fwd | rows_bwd | cols_fwd | cols_bwd]
        const int cbase = (rowp ? 0 : 128) + dir * 64;
        const int half  = lane >> 4;
#pragma unroll
        for (int nt = 0; nt < 4; ++nt)
#pragma unroll
            for (int e = 0; e < 8; ++e) {
                const int m = half * 8 + e;
                out[(size_t)(b0 + m) * 256 + cbase + nt * 16 + nl] = h[nt][e];
            }
    }
}

extern "C" void kernel_launch(void* const* d_in, const int* in_sizes, int n_in,
                              void* d_out, int out_size, void* d_ws, size_t ws_size,
                              hipStream_t stream) {
    const float* inp       = (const float*)d_in[0];
    // d_in[1] = batch_size (int) — B derived from in_sizes[0] instead
    const float* W1        = (const float*)d_in[2];
    const float* U1        = (const float*)d_in[3];
    const float* bg1       = (const float*)d_in[4];
    const float* bu1       = (const float*)d_in[5];
    const float* zeta1     = (const float*)d_in[6];
    const float* nu1       = (const float*)d_in[7];
    const float* W2        = (const float*)d_in[8];
    const float* U2        = (const float*)d_in[9];
    const float* bg2       = (const float*)d_in[10];
    const float* bu2       = (const float*)d_in[11];
    const float* zeta2     = (const float*)d_in[12];
    const float* nu2       = (const float*)d_in[13];
    const float* h0_row    = (const float*)d_in[14];
    const float* h0_col    = (const float*)d_in[15];
    const float* h0_bd_row = (const float*)d_in[16];
    const float* h0_bd_col = (const float*)d_in[17];
    float* out = (float*)d_out;

    const int B = in_sizes[0] / (64 * 8 * 8);          // 16384
    rnnpool_fused<<<B / 16, 512, SMEM_BYTES, stream>>>(
        inp, W1, U1, bg1, bu1, zeta1, nu1,
        W2, U2, bg2, bu2, zeta2, nu2,
        h0_row, h0_col, h0_bd_row, h0_bd_col, out, B);
}
